// AMCNNAttention_32074815766596
// MI455X (gfx1250) — matrix operand order repaired and checked by reading, not compiled, not run on hardware
//
#include <hip/hip_runtime.h>
#include <hip/hip_bf16.h>
#include <cstdint>

// ---------------------------------------------------------------------------
// AMCNN attention for MI455X (gfx1250, wave32, WMMA bf16 16x16x32,
// global_load_async_to_lds_b128 staging)
// B=32, L=1024, D=768, C=2
// ---------------------------------------------------------------------------

typedef __attribute__((ext_vector_type(16))) __bf16 v16bf;
typedef __attribute__((ext_vector_type(8)))  float  v8f;

constexpr int Bc = 32, Lc = 1024, Dc = 768, Cc = 2;
constexpr int Mfull = Bc * Lc;           // 32768
constexpr int BM = 128, BN = 128, BK = 32;
constexpr int LDSK = 40;                 // padded LDS K-stride (bf16 elements)
constexpr int NKT = Dc / BK;             // 24 K-steps

#define DEV __device__ __forceinline__

DEV uint32_t f2bf1(float f) {            // f32 -> bf16 (RNE), low 16 bits
  uint32_t u = __float_as_uint(f);
  uint32_t r = (u + 0x7FFFu + ((u >> 16) & 1u)) >> 16;
  return r & 0xFFFFu;
}
DEV uint32_t f2bf2(float lo, float hi) { return f2bf1(lo) | (f2bf1(hi) << 16); }

DEV float fast_tanh(float x) {           // hardware v_exp_f32 based tanh
  x = fminf(20.f, fmaxf(-20.f, x));
  const float t = __expf(-2.f * x);
  return (1.f - t) / (1.f + t);
}
DEV float fast_sigmoid(float x) { return 1.f / (1.f + __expf(-x)); }

// 32-byte async global->LDS copy: two b128 issues; IOFFSET advances both the
// global and the LDS address (ISA 08_async_tensor §4.4), tracked by ASYNCcnt.
DEV void async_copy32B(uint32_t lds_off, const void* gptr) {
  asm volatile(
      "global_load_async_to_lds_b128 %0, %1, off\n\t"
      "global_load_async_to_lds_b128 %0, %1, off offset:16"
      :: "v"(lds_off), "v"(gptr) : "memory");
}
DEV void wait_async_all() {
  asm volatile("s_wait_asynccnt 0x0" ::: "memory");
}
DEV uint32_t lds_addr(const void* p) {   // generic LDS ptr -> 32-bit LDS offset
  return (uint32_t)(size_t)p;
}

// ---------------------------------------------------------------------------
// Tiled WMMA GEMM:  C_tile[m][n] = sum_k A[m][k] * Bt[n][k]
//   A: bf16 row-major (M x K), Bt: bf16 row-major (N x K)
// MODE 0: store result as bf16 into Uout             (U = X @ w_l^T)
// MODE 1: epilogue sum_n sigmoid(v + b_v[n])*w_v1[n] -> partial per N-tile
// MODE 2: epilogue sum_n tanh(v + b_l)               -> partial per N-tile
// ---------------------------------------------------------------------------
template<int MODE>
__global__ void __launch_bounds__(256)
gemm_wmma_bf16(const unsigned short* __restrict__ Xbf,
               const unsigned short* __restrict__ Bmat,
               unsigned short* __restrict__ Uout,
               float* __restrict__ Pout,
               const float* __restrict__ e0,   // MODE2: b_l   MODE1: b_v
               const float* __restrict__ e1)   // MODE1: w_v1
{
  __shared__ unsigned short As[2][BM * LDSK];
  __shared__ unsigned short Bs[2][BN * LDSK];
  __shared__ float part[4][BM];

  const int z = blockIdx.z;
  int c;
  const unsigned short* Ag;
  const unsigned short* Bg;
  if constexpr (MODE == 2) {
    c = z >> 5;
    const int b = z & 31;
    Ag = Xbf + (size_t)b * Lc * Dc;          // x[b]      (1024 x 768, bf16)
    Bg = Bmat + (size_t)z * Lc * Dc;         // U[c,b]    (1024 x 768, bf16)
  } else {
    c = z;
    Ag = Xbf;                                 // x         (32768 x 768, bf16)
    Bg = Bmat + (size_t)c * Dc * Dc;          // w_l[c] / w_v2t[c] (768x768, bf16)
  }

  const int mTile = blockIdx.y * BM;
  const int nTile = blockIdx.x * BN;

  const int tid  = threadIdx.x;
  const int lane = tid & 31;
  const int wave = tid >> 5;
  const int wm   = (wave >> 2) * 64;          // wave M offset in tile
  const int wn   = (wave & 3) * 32;           // wave N offset in tile
  const int l15  = lane & 15;
  const int lh   = lane >> 4;                 // lane half (0/1)

  const int sRow = tid >> 1;                  // staging row 0..127
  const int sKh  = (tid & 1) * 16;            // staging K half 0/16 (elements)

  const unsigned short* aSrc = Ag + (size_t)(mTile + sRow) * Dc + sKh;
  const unsigned short* bSrc = Bg + (size_t)(nTile + sRow) * Dc + sKh;

  const uint32_t aD0 = lds_addr(&As[0][sRow * LDSK + sKh]);
  const uint32_t aD1 = lds_addr(&As[1][sRow * LDSK + sKh]);
  const uint32_t bD0 = lds_addr(&Bs[0][sRow * LDSK + sKh]);
  const uint32_t bD1 = lds_addr(&Bs[1][sRow * LDSK + sKh]);

  // -------- prologue: async-stage K-step 0 into buffer 0 --------
  async_copy32B(aD0, aSrc);
  async_copy32B(bD0, bSrc);

  v8f acc[4][2];
  const v8f zacc = {0.f, 0.f, 0.f, 0.f, 0.f, 0.f, 0.f, 0.f};
#pragma unroll
  for (int i = 0; i < 4; ++i)
#pragma unroll
    for (int j = 0; j < 2; ++j) acc[i][j] = zacc;

  int buf = 0;
  for (int kt = 0; kt < NKT; ++kt) {
    wait_async_all();          // our async DMAs into current buffer landed
    __syncthreads();           // everyone's landed -> tile readable

    if (kt + 1 < NKT) {        // overlap next tile's DMA with the WMMAs below
      const unsigned short* an = aSrc + (kt + 1) * BK;
      const unsigned short* bn = bSrc + (kt + 1) * BK;
      async_copy32B(buf ? aD0 : aD1, an);
      async_copy32B(buf ? bD0 : bD1, bn);
      if (kt + 2 < NKT) {      // L2 prefetch for K+2
        __builtin_prefetch(aSrc + (kt + 2) * BK, 0, 1);
        __builtin_prefetch(bSrc + (kt + 2) * BK, 0, 1);
      }
    }

    // fragment loads from LDS per ISA 16-bit A(16x32)/B(32x16) layouts
    union Frag { uint4 q[2]; v16bf v; };
    Frag af[4], bfr[2];
#pragma unroll
    for (int i = 0; i < 4; ++i) {
      const unsigned short* p = &As[buf][(wm + i * 16 + l15) * LDSK + lh * 8];
      af[i].q[0] = *reinterpret_cast<const uint4*>(p);        // K = klo..klo+7
      af[i].q[1] = *reinterpret_cast<const uint4*>(p + 16);   // K = klo+16..klo+23
    }
#pragma unroll
    for (int j = 0; j < 2; ++j) {
      const unsigned short* p = &Bs[buf][(wn + j * 16 + l15) * LDSK + lh * 8];
      bfr[j].q[0] = *reinterpret_cast<const uint4*>(p);       // K = klo..klo+7
      bfr[j].q[1] = *reinterpret_cast<const uint4*>(p + 16);  // K = klo+16..klo+23
    }

#pragma unroll
    for (int i = 0; i < 4; ++i)
#pragma unroll
      for (int j = 0; j < 2; ++j)
        acc[i][j] = __builtin_amdgcn_wmma_f32_16x16x32_bf16(
            false, af[i].v, false, bfr[j].v, (short)0, acc[i][j], false, false);

    buf ^= 1;
  }

  // ------------------------------ epilogues --------------------------------
  if constexpr (MODE == 0) {
    // C/D layout: VGPR r -> row r + lh*8, col = l15
#pragma unroll
    for (int i = 0; i < 4; ++i)
#pragma unroll
      for (int j = 0; j < 2; ++j)
#pragma unroll
        for (int r = 0; r < 8; ++r) {
          const int row = mTile + wm + i * 16 + lh * 8 + r;
          const int col = nTile + wn + j * 16 + l15;
          Uout[((size_t)c * Mfull + row) * Dc + col] = (unsigned short)f2bf1(acc[i][j][r]);
        }
  } else {
    float s0 = 0.f, s1 = 0.f, w0 = 0.f, w1 = 0.f;
    if constexpr (MODE == 1) {
      const int n0 = nTile + wn + l15;
      s0 = e0[c * Dc + n0];        s1 = e0[c * Dc + n0 + 16];   // b_v
      w0 = e1[c * Dc + n0];        w1 = e1[c * Dc + n0 + 16];   // w_v1
    } else {
      s0 = e0[c];                                               // b_l
    }
#pragma unroll
    for (int i = 0; i < 4; ++i) {
#pragma unroll
      for (int r = 0; r < 8; ++r) {
        float p;
        if constexpr (MODE == 1) {
          p = fast_sigmoid(acc[i][0][r] + s0) * w0 +
              fast_sigmoid(acc[i][1][r] + s1) * w1;
        } else {
          p = fast_tanh(acc[i][0][r] + s0) + fast_tanh(acc[i][1][r] + s0);
        }
        p += __shfl_xor(p, 8, 32);
        p += __shfl_xor(p, 4, 32);
        p += __shfl_xor(p, 2, 32);
        p += __shfl_xor(p, 1, 32);
        if (l15 == 0) part[wave & 3][wm + i * 16 + lh * 8 + r] = p;
      }
    }
    __syncthreads();
    if (tid < BM) {
      const float v = part[0][tid] + part[1][tid] + part[2][tid] + part[3][tid];
      if constexpr (MODE == 1)
        Pout[((size_t)c * 6 + blockIdx.x) * (size_t)Mfull + mTile + tid] = v;   // pG[c][nt][m]
      else
        Pout[((size_t)z * 8 + blockIdx.x) * (size_t)Lc + mTile + tid] = v;      // pS[cb][nt][i]
    }
  }
}

// --------------------------- prep kernels ----------------------------------
// x (f32) -> bf16, 4 elements / thread
__global__ void convert_x_bf16(const float* __restrict__ in,
                               unsigned short* __restrict__ out) {
  const size_t i4 = (size_t)blockIdx.x * 256 + threadIdx.x;
  const float4 v = reinterpret_cast<const float4*>(in)[i4];
  uint2 p;
  p.x = f2bf2(v.x, v.y);
  p.y = f2bf2(v.z, v.w);
  reinterpret_cast<uint2*>(out)[i4] = p;
}

__global__ void convert_f32_bf16(const float* __restrict__ in,
                                 unsigned short* __restrict__ out, int n) {
  const int i = blockIdx.x * 256 + threadIdx.x;
  if (i < n) out[i] = (unsigned short)f2bf1(in[i]);
}

// out[c][e][d] = in[c][d][e]  (D x D per c), f32 -> bf16
__global__ void transpose_convert(const float* __restrict__ in,
                                  unsigned short* __restrict__ out) {
  __shared__ float t[32][33];
  const int c = blockIdx.z;
  const int d0 = blockIdx.y * 32, e0 = blockIdx.x * 32;
  const int tx = threadIdx.x, ty = threadIdx.y;   // block (32,8)
#pragma unroll
  for (int k = 0; k < 32; k += 8)
    t[ty + k][tx] = in[((size_t)c * Dc + d0 + ty + k) * Dc + e0 + tx];
  __syncthreads();
#pragma unroll
  for (int k = 0; k < 32; k += 8)
    out[((size_t)c * Dc + e0 + ty + k) * Dc + d0 + tx] = (unsigned short)f2bf1(t[tx][ty + k]);
}

// ---------------- softmax over L with partial-tile reduction ---------------
// MODE 2: pS layout [cb][8][1024] ; MODE 1: pG layout [c][6][32768]
template<int MODE>
__global__ void softmax_rows(const float* __restrict__ part, float* __restrict__ out) {
  const int cb = blockIdx.x;
  const int c = cb >> 5, b = cb & 31;
  const int tid = threadIdx.x;
  const int lane = tid & 31, wave = tid >> 5;
  __shared__ float red[8];

  float v[4];
#pragma unroll
  for (int k = 0; k < 4; ++k) {
    const int i = tid + k * 256;
    float s = 0.f;
    if constexpr (MODE == 2) {
#pragma unroll
      for (int t = 0; t < 8; ++t) s += part[((size_t)cb * 8 + t) * Lc + i];
    } else {
#pragma unroll
      for (int t = 0; t < 6; ++t) s += part[((size_t)c * 6 + t) * Mfull + (size_t)b * Lc + i];
    }
    v[k] = s;
  }
  float m = fmaxf(fmaxf(v[0], v[1]), fmaxf(v[2], v[3]));
#pragma unroll
  for (int d = 16; d >= 1; d >>= 1) m = fmaxf(m, __shfl_xor(m, d, 32));
  if (lane == 0) red[wave] = m;
  __syncthreads();
  m = red[0];
#pragma unroll
  for (int w = 1; w < 8; ++w) m = fmaxf(m, red[w]);
  __syncthreads();
  float e[4], s = 0.f;
#pragma unroll
  for (int k = 0; k < 4; ++k) { e[k] = __expf(v[k] - m); s += e[k]; }
#pragma unroll
  for (int d = 16; d >= 1; d >>= 1) s += __shfl_xor(s, d, 32);
  if (lane == 0) red[wave] = s;
  __syncthreads();
  s = 0.f;
#pragma unroll
  for (int w = 0; w < 8; ++w) s += red[w];
  const float inv = 1.f / s;
#pragma unroll
  for (int k = 0; k < 4; ++k) out[(size_t)cb * Lc + tid + k * 256] = e[k] * inv;
}

// pooled[c,b,d] = sum_l a_vec[c,b,l] * x[b,l,d]
__global__ void pooled_kernel(const float* __restrict__ x,
                              const float* __restrict__ avec,
                              float* __restrict__ pooled) {
  const int cb = blockIdx.x;
  const int b = cb & 31;
  const int tid = threadIdx.x;     // 256 threads, 3 d-values each
  float a0 = 0.f, a1 = 0.f, a2 = 0.f;
  const float* av = avec + (size_t)cb * Lc;
  const float* xb = x + (size_t)b * Lc * Dc;
  for (int l = 0; l < Lc; ++l) {
    const float w = av[l];
    const float* xr = xb + (size_t)l * Dc;
    a0 += w * xr[tid];
    a1 += w * xr[tid + 256];
    a2 += w * xr[tid + 512];
  }
  float* p = pooled + (size_t)cb * Dc;
  p[tid] = a0; p[tid + 256] = a1; p[tid + 512] = a2;
}

// out[b,l,d,c] = a[c,b,l] * x[b,l,d] + pooled[c,b,d]
__global__ void final_kernel(const float* __restrict__ x,
                             const float* __restrict__ a,
                             const float* __restrict__ pooled,
                             float* __restrict__ out) {
  const size_t idx = (size_t)blockIdx.x * 256 + threadIdx.x;  // over B*L*D
  const int d = (int)(idx % Dc);
  const size_t bl = idx / Dc;                                 // b*L + l
  const int b = (int)(bl >> 10);
  const float xv = x[idx];
  const float a0 = a[bl];
  const float a1 = a[(size_t)Mfull + bl];
  const float p0 = pooled[(size_t)b * Dc + d];
  const float p1 = pooled[((size_t)Bc + b) * Dc + d];
  float2 o; o.x = a0 * xv + p0; o.y = a1 * xv + p1;
  reinterpret_cast<float2*>(out)[idx] = o;
}

// ---------------------------------------------------------------------------
extern "C" void kernel_launch(void* const* d_in, const int* in_sizes, int n_in,
                              void* d_out, int out_size, void* d_ws, size_t ws_size,
                              hipStream_t stream) {
  const float* x    = (const float*)d_in[0];
  const float* w_l  = (const float*)d_in[1];
  const float* b_l  = (const float*)d_in[2];
  const float* w_v1 = (const float*)d_in[3];
  const float* w_v2 = (const float*)d_in[4];
  const float* b_v  = (const float*)d_in[5];
  float* out = (float*)d_out;

  char* w = (char*)d_ws;
  unsigned short* Xbf  = (unsigned short*)w; w += (size_t)Mfull * Dc * 2;       // 48 MB
  unsigned short* U    = (unsigned short*)w; w += (size_t)Cc * Mfull * Dc * 2;  // 96 MB
  unsigned short* WlBf = (unsigned short*)w; w += (size_t)Cc * Dc * Dc * 2;
  unsigned short* Wv2t = (unsigned short*)w; w += (size_t)Cc * Dc * Dc * 2;
  float* pS   = (float*)w;   w += (size_t)Cc * Bc * 8 * Lc * 4;
  float* pG   = (float*)w;   w += (size_t)Cc * 6 * Mfull * 4;
  float* aBuf = (float*)w;   w += (size_t)Cc * Mfull * 4;
  float* avec = (float*)w;   w += (size_t)Cc * Mfull * 4;
  float* pooled = (float*)w; w += (size_t)Cc * Bc * Dc * 4;

  // prep: bf16 X and weights (w_l as-is; w_v2 transposed -> both GEMMs "NT")
  convert_x_bf16<<<(Mfull * (Dc / 4)) / 256, 256, 0, stream>>>(x, Xbf);
  {
    const int n = Cc * Dc * Dc;
    convert_f32_bf16<<<(n + 255) / 256, 256, 0, stream>>>(w_l, WlBf, n);
    transpose_convert<<<dim3(Dc / 32, Dc / 32, Cc), dim3(32, 8), 0, stream>>>(w_v2, Wv2t);
  }
  // U[c] = X @ w_l[c]^T   (bf16 out)
  gemm_wmma_bf16<0><<<dim3(Dc / BN, Mfull / BM, Cc), 256, 0, stream>>>(
      Xbf, WlBf, U, nullptr, nullptr, nullptr);
  // vector-attention scores (fused sigmoid + dot(w_v1)), per-N-tile partials
  gemm_wmma_bf16<1><<<dim3(Dc / BN, Mfull / BM, Cc), 256, 0, stream>>>(
      Xbf, Wv2t, nullptr, pG, b_v, w_v1);
  // scalar-attention: S = X_b @ U[c,b]^T, fused tanh + row-sum partials
  gemm_wmma_bf16<2><<<dim3(Lc / BN, Lc / BM, Cc * Bc), 256, 0, stream>>>(
      Xbf, U, nullptr, pS, b_l, nullptr);
  // softmaxes (reduce partials in fixed order -> deterministic)
  softmax_rows<2><<<Cc * Bc, 256, 0, stream>>>(pS, aBuf);
  softmax_rows<1><<<Cc * Bc, 256, 0, stream>>>(pG, avec);
  // pooled vectors
  pooled_kernel<<<Cc * Bc, 256, 0, stream>>>(x, avec, pooled);
  // final combine (B,L,D,C), float2 stores over the C-pairs
  final_kernel<<<(Bc * Lc * Dc) / 256, 256, 0, stream>>>(x, aBuf, pooled, out);

  (void)in_sizes; (void)n_in; (void)out_size; (void)ws_size;
}
